// WindowNeighborhoodMHA2d_52518860096038
// MI455X (gfx1250) — matrix operand order repaired and checked by reading, not compiled, not run on hardware
//
#include <hip/hip_runtime.h>
#include <hip/hip_bf16.h>

typedef __attribute__((ext_vector_type(16))) _Float16 v16h;
typedef __attribute__((ext_vector_type(8)))  float    v8f;

#define C_DIM 256
#define NH    8
#define HD    32
#define NTOK  64
#define IMG_H 128
#define IMG_W 128

// LDS strides (elements). All even (32-bit paired f16 loads stay 4B aligned) and
// chosen so row-stride-in-dwords mod 64 spreads 16 consecutive rows over distinct banks.
#define LDT   264   // t:   [64][256] f16
#define LDQK  40    // qh/kh: [64][32] f16
#define LDVT  72    // vhT: [32][64] f16 (dim-major)
#define LDSC  68    // sc:  [64][64] f32
#define LDP   72    // p:   [64][64] f16
#define LDO   264   // o:   [64][256] f16
#define LDW   264   // wsq: [96][256] f16 staged QKV weight slice

// LDS byte offsets (16B aligned). Total kept under 160KB so 2 workgroups
// co-reside per WGP (320KB LDS) -> 4 waves/SIMD to hide barrier/softmax bubbles.
#define OFF_T   0
#define OFF_QH  (OFF_T  + 64*LDT*2)
#define OFF_KH  (OFF_QH + 64*LDQK*2)
#define OFF_VT  (OFF_KH + 64*LDQK*2)
#define OFF_SC  (OFF_VT + 32*LDVT*2)
#define OFF_P   (OFF_SC + 64*LDSC*4)
#define OFF_O   (OFF_P  + 64*LDP*2)
#define OFF_WSQ (OFF_O  + 64*LDO*2)
#define SMEM_BYTES (OFF_WSQ + 96*LDW*2)   // 159744 B <= 160KB

// ---- CDNA5 async copy: global -> LDS, 16B per lane, tracked by ASYNCcnt ----
__device__ __forceinline__ void async_ld_b128(void* lds, const void* g) {
    unsigned loff = (unsigned)(unsigned long long)lds;  // low 32b of generic LDS ptr = LDS byte addr
    asm volatile("global_load_async_to_lds_b128 %0, %1, off"
                 :: "v"(loff), "v"((unsigned long long)g) : "memory");
}
__device__ __forceinline__ void wait_async0() {
    asm volatile("s_wait_asynccnt 0x0" ::: "memory");
}

// ---- WMMA fragment helpers (A/B: k = 16*(lane>>4) + element, same order in both; C/D per CDNA5 wave32 layout) ----

// A 16x32 f16 from row-major [M][K] (stride ld), tile (m0,k0).
__device__ __forceinline__ v16h load_A16x32(const _Float16* base, int ld, int m0, int k0, int lane) {
    union { v16h h; unsigned u[8]; } r;
    const _Float16* p = base + (m0 + (lane & 15)) * ld + k0 + ((lane >> 4) << 4);
#pragma unroll
    for (int i = 0; i < 8; ++i) r.u[i] = *(const unsigned*)(p + 2 * i);
    return r.h;
}

// B 32x16 f16 sourced from transposed storage Bt[n][k] (stride ld): Bt[n][k] = B[k][n].
__device__ __forceinline__ v16h load_Bt32x16(const _Float16* base, int ld, int k0, int n0, int lane) {
    union { v16h h; unsigned u[8]; } r;
    const _Float16* p = base + (n0 + (lane & 15)) * ld + k0 + ((lane >> 4) << 4);
#pragma unroll
    for (int i = 0; i < 8; ++i) r.u[i] = *(const unsigned*)(p + 2 * i);
    return r.h;
}

// C/D 16x16 f32: VGPR r -> row m0 + r + 8*(lane>=16), col n0 + (lane&15).
__device__ __forceinline__ void store_C_f16(_Float16* base, int ld, int m0, int n0, int lane, v8f c) {
    int col = n0 + (lane & 15);
    int rb  = m0 + ((lane >> 4) << 3);
#pragma unroll
    for (int r = 0; r < 8; ++r) base[(rb + r) * ld + col] = (_Float16)c[r];
}
__device__ __forceinline__ void store_C_f16_T(_Float16* base, int ld, int m0, int n0, int lane, v8f c) {
    int col = n0 + (lane & 15);
    int rb  = m0 + ((lane >> 4) << 3);
#pragma unroll
    for (int r = 0; r < 8; ++r) base[col * ld + rb + r] = (_Float16)c[r];
}

#define WMMA_F16(a, b, c) \
    __builtin_amdgcn_wmma_f32_16x16x32_f16(false, (a), false, (b), (short)0, (c), false, false)

// ---- Prep: weights -> f16 transposed [out][in]; bias/mask table -> global (L2-hot) ----
__global__ void prep_weights_kernel(const float* __restrict__ qkv_w,   // [256][768]
                                    const float* __restrict__ head_w,  // [256][256]
                                    _Float16* __restrict__ wT_qkv,     // [768][256]
                                    _Float16* __restrict__ wT_head,    // [256][256]
                                    float* __restrict__ bm_g) {        // [64][64]
    int idx = blockIdx.x * blockDim.x + threadIdx.x;
    if (idx < 256 * 768) {
        int k = idx / 768, n = idx % 768;
        wT_qkv[n * 256 + k] = (_Float16)qkv_w[idx];
    }
    if (idx < 256 * 256) {
        int k = idx / 256, n = idx % 256;
        wT_head[n * 256 + k] = (_Float16)head_w[idx];
    }
    if (idx < NTOK * NTOK) {
        int qi = idx >> 6, kj = idx & 63;
        float dh = (float)((qi >> 3) - (kj >> 3));
        float dw = (float)((qi & 7) - (kj & 7));
        float d2 = dh * dh + dw * dw;
        bm_g[idx] = (d2 <= 12.25f) ? (-__logf(sqrtf(d2) + 1.0f)) : -1e30f;
    }
}

// ---- Main: one workgroup (8 wave32) per 8x8 window ----
__global__ void __launch_bounds__(256, 1)
win_mha_kernel(const float* __restrict__ x,
               const float* __restrict__ qkv_b,
               const float* __restrict__ head_b,
               const _Float16* __restrict__ wT_qkv,
               const _Float16* __restrict__ wT_head,
               const float* __restrict__ bm_g,
               float* __restrict__ out) {
    extern __shared__ char smem_raw[];
    _Float16* t_   = (_Float16*)(smem_raw + OFF_T);
    _Float16* qh_  = (_Float16*)(smem_raw + OFF_QH);
    _Float16* kh_  = (_Float16*)(smem_raw + OFF_KH);
    _Float16* vT_  = (_Float16*)(smem_raw + OFF_VT);
    float*    sc_  = (float*)   (smem_raw + OFF_SC);
    _Float16* p_   = (_Float16*)(smem_raw + OFF_P);
    _Float16* o_   = (_Float16*)(smem_raw + OFF_O);
    _Float16* wsq_ = (_Float16*)(smem_raw + OFF_WSQ);

    const int tid  = threadIdx.x;
    const int lane = tid & 31;
    const int wv   = tid >> 5;
    const int bid  = blockIdx.x;
    const int gw = bid & 15, gh = (bid >> 4) & 15, b = bid >> 8;

    const float* xwin = x + (long)b * C_DIM * IMG_H * IMG_W + (long)(gh * 8) * IMG_W + gw * 8;

    // Stage window tile -> LDS f16 (float4 per lane: 16B coalesced bursts within window rows).
    for (int idx = tid; idx < C_DIM * 16; idx += 256) {
        int c  = idx >> 4;
        int m4 = (idx & 15) << 2;   // token, multiple of 4
        const float* src = xwin + (long)c * IMG_H * IMG_W + (m4 >> 3) * IMG_W + (m4 & 7);
        float4 v4 = *(const float4*)src;
        t_[(m4 + 0) * LDT + c] = (_Float16)v4.x;
        t_[(m4 + 1) * LDT + c] = (_Float16)v4.y;
        t_[(m4 + 2) * LDT + c] = (_Float16)v4.z;
        t_[(m4 + 3) * LDT + c] = (_Float16)v4.w;
    }
    __syncthreads();

    const float scale = 0.17677669529663689f; // 1/sqrt(32)

    for (int h = 0; h < NH; ++h) {
        // ---- Async-stage this head's QKV weight slice: 96 rows x 256 f16, coalesced 16B/lane ----
        for (int row = wv; row < 96; row += 8) {
            int mat = row >> 5, r32 = row & 31;
            const _Float16* gsrc = wT_qkv + ((long)(mat * C_DIM + h * HD + r32)) * 256 + lane * 8;
            async_ld_b128(wsq_ + row * LDW + lane * 8, gsrc);
        }
        wait_async0();
        __syncthreads();

        // ---- QKV: each wave owns (mt,nt); A fragment shared across Q/K/V accumulators ----
        {
            int m0 = (wv >> 1) << 4;
            int n0 = (wv & 1) << 4;
            v8f ac[3];
#pragma unroll
            for (int mat = 0; mat < 3; ++mat) {
                float bias = qkv_b[mat * C_DIM + h * HD + n0 + (lane & 15)];
#pragma unroll
                for (int r = 0; r < 8; ++r) ac[mat][r] = bias;
            }
#pragma unroll
            for (int kk = 0; kk < 8; ++kk) {
                v16h a = load_A16x32(t_, LDT, m0, kk * 32, lane);
#pragma unroll
                for (int mat = 0; mat < 3; ++mat) {
                    v16h bf = load_Bt32x16(wsq_ + (mat * 32) * LDW, LDW, kk * 32, n0, lane);
                    ac[mat] = WMMA_F16(a, bf, ac[mat]);
                }
            }
            store_C_f16  (qh_, LDQK, m0, n0, lane, ac[0]);
            store_C_f16  (kh_, LDQK, m0, n0, lane, ac[1]);
            store_C_f16_T(vT_, LDVT, m0, n0, lane, ac[2]);  // V^T [dim][token]
        }
        __syncthreads();

        // ---- Scores S = Q K^T: A fragment reused across 2 N-tiles per wave ----
        {
            int m0 = (wv >> 1) << 4;
            v16h a = load_A16x32(qh_, LDQK, m0, 0, lane);
#pragma unroll
            for (int t2 = 0; t2 < 2; ++t2) {
                int n0 = (((wv & 1) << 1) + t2) << 4;
                v8f acc;
#pragma unroll
                for (int r = 0; r < 8; ++r) acc[r] = 0.0f;
                v16h bf = load_Bt32x16(kh_, LDQK, 0, n0, lane);
                acc = WMMA_F16(a, bf, acc);
                int col = n0 + (lane & 15);
                int rb  = m0 + ((lane >> 4) << 3);
#pragma unroll
                for (int r = 0; r < 8; ++r)
                    sc_[(rb + r) * LDSC + col] = acc[r] * scale + bm_g[(rb + r) * 64 + col];
            }
        }
        __syncthreads();

        // ---- Softmax: 4 lanes per row, shfl_xor reductions; row segment cached in regs ----
        {
            int row = tid >> 2;
            int seg = (tid & 3) << 4;
            float ev[16];
            float mx = -1e30f;
#pragma unroll
            for (int j2 = 0; j2 < 16; ++j2) {
                ev[j2] = sc_[row * LDSC + seg + j2];
                mx = fmaxf(mx, ev[j2]);
            }
            mx = fmaxf(mx, __shfl_xor(mx, 1, 32));
            mx = fmaxf(mx, __shfl_xor(mx, 2, 32));
            float sum = 0.0f;
#pragma unroll
            for (int j2 = 0; j2 < 16; ++j2) {
                ev[j2] = __expf(ev[j2] - mx);
                sum += ev[j2];
            }
            sum += __shfl_xor(sum, 1, 32);
            sum += __shfl_xor(sum, 2, 32);
            float inv = 1.0f / sum;
#pragma unroll
            for (int j2 = 0; j2 < 16; ++j2)
                p_[row * LDP + seg + j2] = (_Float16)(ev[j2] * inv);
        }
        __syncthreads();

        // ---- O_h = P @ V: one tile per wave, K=64 (2 steps) ----
        {
            int m0 = (wv >> 1) << 4;
            int n0 = (wv & 1) << 4;
            v8f acc;
#pragma unroll
            for (int r = 0; r < 8; ++r) acc[r] = 0.0f;
#pragma unroll
            for (int kk = 0; kk < 2; ++kk) {
                v16h a  = load_A16x32(p_, LDP, m0, kk * 32, lane);
                v16h bf = load_Bt32x16(vT_, LDVT, kk * 32, n0, lane);
                acc = WMMA_F16(a, bf, acc);
            }
            store_C_f16(o_, LDO, m0, h * HD + n0, lane, acc);
        }
        __syncthreads();
    }

    // ---- Output projection: each wave owns 1 M-tile x 8 N-tiles (A reused 8x) ----
    {
        int m0    = (wv >> 1) << 4;
        int nbase = (wv & 1) << 7;   // 0 or 128
        v8f ac[8];
#pragma unroll
        for (int i = 0; i < 8; ++i) {
            float bias = head_b[nbase + i * 16 + (lane & 15)];
#pragma unroll
            for (int r = 0; r < 8; ++r) ac[i][r] = bias;
        }
#pragma unroll
        for (int kk = 0; kk < 8; ++kk) {
            v16h a = load_A16x32(o_, LDO, m0, kk * 32, lane);
#pragma unroll
            for (int i = 0; i < 8; ++i) {
                v16h bf = load_Bt32x16(wT_head, 256, kk * 32, nbase + i * 16, lane);
                ac[i] = WMMA_F16(a, bf, ac[i]);
            }
        }
        // Each lane's 8 rows are tokens rb..rb+7 = one window row -> 32B contiguous: 2x float4.
        int rb = m0 + ((lane >> 4) << 3);
#pragma unroll
        for (int i = 0; i < 8; ++i) {
            int c = nbase + i * 16 + (lane & 15);
            float* op = out + (long)b * C_DIM * IMG_H * IMG_W + (long)c * IMG_H * IMG_W
                            + (long)(gh * 8 + (rb >> 3)) * IMG_W + gw * 8;
            float4 lo, hi;
            lo.x = ac[i][0]; lo.y = ac[i][1]; lo.z = ac[i][2]; lo.w = ac[i][3];
            hi.x = ac[i][4]; hi.y = ac[i][5]; hi.z = ac[i][6]; hi.w = ac[i][7];
            *(float4*)(op)     = lo;
            *(float4*)(op + 4) = hi;
        }
    }
}

extern "C" void kernel_launch(void* const* d_in, const int* in_sizes, int n_in,
                              void* d_out, int out_size, void* d_ws, size_t ws_size,
                              hipStream_t stream) {
    const float* x      = (const float*)d_in[0];
    const float* qkv_w  = (const float*)d_in[1];
    const float* qkv_b  = (const float*)d_in[2];
    const float* head_w = (const float*)d_in[3];
    const float* head_b = (const float*)d_in[4];
    float* out = (float*)d_out;

    _Float16* wT_qkv  = (_Float16*)d_ws;                      // 768*256 f16
    _Float16* wT_head = wT_qkv + 768 * 256;                   // 256*256 f16
    float*    bm_g    = (float*)(wT_head + 256 * 256);        // 64*64 f32 (shared, L2-hot)

    prep_weights_kernel<<<768, 256, 0, stream>>>(qkv_w, head_w, wT_qkv, wT_head, bm_g);

    (void)hipFuncSetAttribute((const void*)win_mha_kernel,
                              hipFuncAttributeMaxDynamicSharedMemorySize, SMEM_BYTES);

    win_mha_kernel<<<2048, 256, SMEM_BYTES, stream>>>(x, qkv_b, head_b,
                                                      wT_qkv, wT_head, bm_g, out);
}
